// SelfAttention_61366492725630
// MI455X (gfx1250) — matrix operand order repaired and checked by reading, not compiled, not run on hardware
//
#include <hip/hip_runtime.h>
#include <hip/hip_bf16.h>

// Self-attention (B=4, T=2048, K=128, H=8) fused flash-attention pipeline for
// gfx1250: bf16 WMMA everywhere, f32 accumulate, double-buffered async LDS
// staging of K/V (ASYNCcnt in-order drain => software pipeline).

#define Bb 4
#define Tt 2048
#define Kk 128
#define Hh 8
#define HK (Hh * Kk)   // 1024
#define BT (Bb * Tt)   // 8192

typedef __bf16 bf16_t;
typedef __attribute__((ext_vector_type(16))) __bf16 v16bf;
typedef __attribute__((ext_vector_type(8)))  __bf16 bf16x8;
typedef __attribute__((ext_vector_type(8)))  float  v8f;

union Frag {
  v16bf  v;
  bf16x8 h[2];
};

// ---- WMMA fragment loaders (wave32 layouts per CDNA5 ISA 7.12.2) ----------
// A (16x32 bf16, MxK, row-major src, K contiguous):
//   lane L<16 : row M=L,    K = [0..7]  and [16..23]
//   lane L>=16: row M=L-16, K = [8..15] and [24..31]
__device__ inline v16bf load_a_frag(const bf16_t* __restrict__ src, int ld) {
  const int lane = threadIdx.x & 31;
  const int row  = lane & 15;
  const int half = lane >> 4;
  const bf16_t* p = src + (size_t)row * ld + 8 * half;
  Frag f;
  f.h[0] = *(const bf16x8*)(p);
  f.h[1] = *(const bf16x8*)(p + 16);
  return f.v;
}

// B (32x16 bf16, KxN) from N-major storage (each column = contiguous K run):
//   lane L<16 : col N=L, K = [0..7] and [16..23];  lane L>=16: col N=L-16, K = [8..15] and [24..31]
__device__ inline v16bf load_b_frag(const bf16_t* __restrict__ srcT, int ld) {
  const int lane = threadIdx.x & 31;
  const int col  = lane & 15;
  const int half = lane >> 4;
  const bf16_t* p = srcT + (size_t)col * ld + 8 * half;
  Frag f;
  f.h[0] = *(const bf16x8*)(p);
  f.h[1] = *(const bf16x8*)(p + 16);
  return f.v;
}

__device__ inline v8f wmma_bf16(v16bf a, v16bf b, v8f c) {
  return __builtin_amdgcn_wmma_f32_16x16x32_bf16(
      /*neg_a=*/false, a, /*neg_b=*/false, b,
      /*c_mod=*/(short)0, c, /*reuse_a=*/false, /*reuse_b=*/false);
}

// ---- CDNA5 async memory->LDS copy (ASYNCcnt-tracked, per-lane B128) -------
__device__ inline void async_load_b128(unsigned lds_byte_off,
                                       const bf16_t* gaddr) {
  asm volatile("global_load_async_to_lds_b128 %0, %1, off"
               :: "v"(lds_byte_off), "v"(gaddr) : "memory");
}
__device__ inline void wait_async0() {                 // drain everything
  asm volatile("s_wait_asynccnt 0" ::: "memory");
}
__device__ inline void wait_async8() {                 // keep next tile (8 ops)
  asm volatile("s_wait_asynccnt 8" ::: "memory");      // in flight
}

// ---- Precision-conversion / transpose kernels -----------------------------
__global__ void convert_f32_bf16_kernel(const float* __restrict__ src,
                                        bf16_t* __restrict__ dst, int n) {
  int i = blockIdx.x * blockDim.x + threadIdx.x;
  if (i < n) dst[i] = (bf16_t)src[i];
}

// src [R,C] f32 row-major -> dst [C,R] bf16 row-major (so K becomes contiguous)
__global__ void transpose_convert_kernel(const float* __restrict__ src,
                                         bf16_t* __restrict__ dst, int R, int C) {
  int i = blockIdx.x * blockDim.x + threadIdx.x;
  if (i < R * C) {
    int r = i / C, c = i % C;
    dst[(size_t)c * R + r] = (bf16_t)src[i];
  }
}

// ---- QKV projection: [BT,128] @ [128,1024], one wave per 16x16 tile -------
// q_ws,k_ws: [B,H,T,128] bf16 (scaled by 128^-0.25). v_ws: [B,H,128,T] bf16
// (transposed so PV B-fragments are contiguous).
__global__ void qkv_proj_kernel(const bf16_t* __restrict__ Xb,
                                const bf16_t* __restrict__ WqT,
                                const bf16_t* __restrict__ WkT,
                                const bf16_t* __restrict__ WvT,
                                bf16_t* __restrict__ q_ws,
                                bf16_t* __restrict__ k_ws,
                                bf16_t* __restrict__ v_ws) {
  const int rt    = blockIdx.x;   // row tile over BT
  const int ct    = blockIdx.y;   // col tile over 1024
  const int which = blockIdx.z;   // 0=Q 1=K 2=V
  const bf16_t* WT = (which == 0) ? WqT : (which == 1) ? WkT : WvT;

  v8f acc = {};
#pragma unroll
  for (int k0 = 0; k0 < Kk; k0 += 32) {
    v16bf a = load_a_frag(Xb + (size_t)rt * 16 * Kk + k0, Kk);
    v16bf b = load_b_frag(WT + (size_t)ct * 16 * Kk + k0, Kk);
    acc = wmma_bf16(a, b, acc);
  }

  const int lane = threadIdx.x & 31;
  const int col  = lane & 15;
  const int half = lane >> 4;
  const int n = ct * 16 + col;        // 0..1023
  const int h = n >> 7;
  const int d = n & (Kk - 1);
  const float scale = (which == 2) ? 1.0f : 0.29730177875068026f; // 128^-0.25
#pragma unroll
  for (int r = 0; r < 8; ++r) {
    int m  = rt * 16 + r + 8 * half;  // global row over BT
    int bb = m >> 11;                 // /T
    int t  = m & (Tt - 1);
    bf16_t val = (bf16_t)(acc[r] * scale);
    size_t bh = (size_t)bb * Hh + h;
    if (which == 0)      q_ws[(bh * Tt + t) * Kk + d] = val;
    else if (which == 1) k_ws[(bh * Tt + t) * Kk + d] = val;
    else                 v_ws[(bh * Kk + d) * Tt + t] = val;
  }
}

// ---- Flash attention: 4 waves/block; 64 query rows; double-buffered -------
__global__ void __launch_bounds__(128)
flash_attn_kernel(const bf16_t* __restrict__ q_ws,
                  const bf16_t* __restrict__ k_ws,
                  const bf16_t* __restrict__ v_ws,
                  bf16_t* __restrict__ o_ws) {
  __shared__ __attribute__((aligned(16))) bf16_t k_lds[2][32 * Kk];    // 16KB
  __shared__ __attribute__((aligned(16))) bf16_t v_lds[2][Kk * 32];    // 16KB
  __shared__ __attribute__((aligned(16))) bf16_t p_lds[4][16 * 32];    //  4KB

  const int tid  = threadIdx.x;
  const int wave = tid >> 5;
  const int lane = tid & 31;
  const int col  = lane & 15;
  const int half = lane >> 4;

  const int qt = blockIdx.x;  // 64-row query tile (T/64)
  const int h  = blockIdx.y;
  const int b  = blockIdx.z;
  const size_t bh = (size_t)b * Hh + h;
  const bf16_t* Qw = q_ws + bh * Tt * Kk + ((size_t)qt * 64 + wave * 16) * Kk;
  const bf16_t* Km = k_ws + bh * Tt * Kk;
  const bf16_t* Vt = v_ws + bh * (size_t)Kk * Tt;

  const unsigned k_base = (unsigned)(uintptr_t)(&k_lds[0][0]);
  const unsigned v_base = (unsigned)(uintptr_t)(&v_lds[0][0]);

  // Stage one 32-key K/V tile into LDS buffer `buf` (8 async B128 per thread).
  auto stage = [&](int buf, int s0) {
    const unsigned kb = k_base + (unsigned)buf * (32 * Kk * 2);
    const unsigned vb = v_base + (unsigned)buf * (Kk * 32 * 2);
#pragma unroll
    for (int c = 0; c < 4; ++c) {
      int chunk = c * 128 + tid;         // 0..511, lane-contiguous
      int row = chunk >> 4;              // key row
      int e8  = (chunk & 15) * 8;        // element offset within row
      async_load_b128(kb + (unsigned)(row * Kk + e8) * 2,
                      Km + (size_t)(s0 + row) * Kk + e8);
    }
#pragma unroll
    for (int c = 0; c < 4; ++c) {
      int chunk = c * 128 + tid;         // 0..511
      int d  = chunk >> 2;               // d row
      int e8 = (chunk & 3) * 8;          // key offset within row
      async_load_b128(vb + (unsigned)(d * 32 + e8) * 2,
                      Vt + (size_t)d * Tt + s0 + e8);
    }
  };

  // Q fragments for this wave's 16 rows: 16x128 = 4 x (16x32)
  v16bf qf[4];
#pragma unroll
  for (int i = 0; i < 4; ++i) qf[i] = load_a_frag(Qw + i * 32, Kk);

  v8f acc[8];                 // O accumulator 16x128
  float m_i[8], l_i[8];       // online-softmax row stats (rows r+8*half)
#pragma unroll
  for (int dt = 0; dt < 8; ++dt) acc[dt] = {};
#pragma unroll
  for (int r = 0; r < 8; ++r) { m_i[r] = -1e30f; l_i[r] = 0.0f; }

  stage(0, 0);  // prologue: prefetch first tile

  for (int s0 = 0; s0 < Tt; s0 += 32) {
    const int buf = (s0 >> 5) & 1;
    const bool has_next = (s0 + 32) < Tt;
    if (has_next) {
      stage(buf ^ 1, s0 + 32);  // issue next tile behind current compute
      wait_async8();            // in-order drain: current tile landed,
    } else {                    // next tile (8 newest ops) still in flight
      wait_async0();
    }
    __syncthreads();            // current tile visible to all 4 waves

    // --- S = Q K^T for 32 keys (from LDS): two 16x16 n-tiles, 4 d-chunks ---
    const bf16_t* kbuf = &k_lds[buf][0];
    const bf16_t* vbuf = &v_lds[buf][0];
    v8f sf[2];
#pragma unroll
    for (int nt = 0; nt < 2; ++nt) {
      v8f s = {};
#pragma unroll
      for (int dk = 0; dk < 4; ++dk) {
        v16bf kf = load_b_frag(kbuf + (nt * 16) * Kk + dk * 32, Kk);
        s = wmma_bf16(qf[dk], kf, s);
      }
      sf[nt] = s;
    }

    // --- online softmax; stage P (bf16) through per-wave LDS ---
    float corr[8];
#pragma unroll
    for (int r = 0; r < 8; ++r) {
      float mx = fmaxf(sf[0][r], sf[1][r]);
#pragma unroll
      for (int off = 1; off < 16; off <<= 1)
        mx = fmaxf(mx, __shfl_xor(mx, off, 32));  // reduce within 16-lane half
      float m_new = fmaxf(m_i[r], mx);
      corr[r] = __expf(m_i[r] - m_new);
      float p0 = __expf(sf[0][r] - m_new);
      float p1 = __expf(sf[1][r] - m_new);
      p_lds[wave][(r + 8 * half) * 32 + col]      = (bf16_t)p0;
      p_lds[wave][(r + 8 * half) * 32 + 16 + col] = (bf16_t)p1;
      float ps = p0 + p1;
#pragma unroll
      for (int off = 1; off < 16; off <<= 1)
        ps += __shfl_xor(ps, off, 32);
      l_i[r] = l_i[r] * corr[r] + ps;
      m_i[r] = m_new;
    }
#pragma unroll
    for (int dt = 0; dt < 8; ++dt)
#pragma unroll
      for (int r = 0; r < 8; ++r)
        acc[dt][r] *= corr[r];

    // --- O += P V (V^T tile from LDS) ---
    v16bf pf = load_a_frag(&p_lds[wave][0], 32);  // same-wave LDS RAW: in-order
#pragma unroll
    for (int dt = 0; dt < 8; ++dt) {
      v16bf vf = load_b_frag(vbuf + (dt * 16) * 32, 32);
      acc[dt] = wmma_bf16(pf, vf, acc[dt]);
    }
    __syncthreads();  // all waves done with buf before it is restaged (i+2)
  }

  // Epilogue: normalize and emit to [B, T, H*K] bf16 for the output GEMM.
#pragma unroll
  for (int dt = 0; dt < 8; ++dt) {
#pragma unroll
    for (int r = 0; r < 8; ++r) {
      int t = qt * 64 + wave * 16 + r + 8 * half;
      int c = h * Kk + dt * 16 + col;
      float o = acc[dt][r] / l_i[r];
      o_ws[((size_t)b * Tt + t) * HK + c] = (bf16_t)o;
    }
  }
}

// ---- Output projection: [BT,1024] @ [1024,128] + bias ---------------------
__global__ void out_proj_kernel(const bf16_t* __restrict__ Ob,
                                const bf16_t* __restrict__ WuT,
                                const float* __restrict__ bu,
                                float* __restrict__ out) {
  const int rt = blockIdx.x;  // row tile over BT
  const int ct = blockIdx.y;  // col tile over 128
  const bf16_t* Abase = Ob  + (size_t)rt * 16 * HK;
  const bf16_t* Bbase = WuT + (size_t)ct * 16 * HK;
  v8f acc = {};
#pragma unroll 4
  for (int k0 = 0; k0 < HK; k0 += 32) {
    __builtin_prefetch(Abase + k0 + 128, 0, 3);   // global_prefetch_b8
    __builtin_prefetch(Bbase + k0 + 128, 0, 3);
    v16bf a = load_a_frag(Abase + k0, HK);
    v16bf b = load_b_frag(Bbase + k0, HK);
    acc = wmma_bf16(a, b, acc);
  }
  const int lane = threadIdx.x & 31;
  const int col  = lane & 15;
  const int half = lane >> 4;
#pragma unroll
  for (int r = 0; r < 8; ++r) {
    int row = rt * 16 + r + 8 * half;
    int c   = ct * 16 + col;
    out[(size_t)row * Kk + c] = acc[r] + bu[c];
  }
}

// ---- Host launch ----------------------------------------------------------
extern "C" void kernel_launch(void* const* d_in, const int* in_sizes, int n_in,
                              void* d_out, int out_size, void* d_ws, size_t ws_size,
                              hipStream_t stream) {
  const float* x  = (const float*)d_in[0];
  const float* Wq = (const float*)d_in[1];
  const float* Wk = (const float*)d_in[2];
  const float* Wv = (const float*)d_in[3];
  const float* Wu = (const float*)d_in[4];
  const float* bu = (const float*)d_in[5];
  float* out = (float*)d_out;

  char* wsb = (char*)d_ws;
  size_t off = 0;
  auto alloc = [&](size_t bytes) -> void* {
    void* p = wsb + off;
    off += (bytes + 255) & ~(size_t)255;
    return p;
  };
  bf16_t* Xb   = (bf16_t*)alloc((size_t)BT * Kk * sizeof(bf16_t));   //  2 MB
  bf16_t* WqT  = (bf16_t*)alloc((size_t)HK * Kk * sizeof(bf16_t));
  bf16_t* WkT  = (bf16_t*)alloc((size_t)HK * Kk * sizeof(bf16_t));
  bf16_t* WvT  = (bf16_t*)alloc((size_t)HK * Kk * sizeof(bf16_t));
  bf16_t* WuT  = (bf16_t*)alloc((size_t)Kk * HK * sizeof(bf16_t));
  bf16_t* q_ws = (bf16_t*)alloc((size_t)Bb * Hh * Tt * Kk * sizeof(bf16_t)); // 16 MB
  bf16_t* k_ws = (bf16_t*)alloc((size_t)Bb * Hh * Tt * Kk * sizeof(bf16_t)); // 16 MB
  bf16_t* v_ws = (bf16_t*)alloc((size_t)Bb * Hh * Kk * Tt * sizeof(bf16_t)); // 16 MB
  bf16_t* o_ws = (bf16_t*)alloc((size_t)BT * HK * sizeof(bf16_t));           // 16 MB

  // 1) conversions
  {
    int n = BT * Kk;
    convert_f32_bf16_kernel<<<(n + 255) / 256, 256, 0, stream>>>(x, Xb, n);
    int nw = Kk * HK;
    transpose_convert_kernel<<<(nw + 255) / 256, 256, 0, stream>>>(Wq, WqT, Kk, HK);
    transpose_convert_kernel<<<(nw + 255) / 256, 256, 0, stream>>>(Wk, WkT, Kk, HK);
    transpose_convert_kernel<<<(nw + 255) / 256, 256, 0, stream>>>(Wv, WvT, Kk, HK);
    transpose_convert_kernel<<<(nw + 255) / 256, 256, 0, stream>>>(Wu, WuT, HK, Kk);
  }
  // 2) QKV projections (WMMA)
  {
    dim3 grid(BT / 16, HK / 16, 3);
    qkv_proj_kernel<<<grid, 32, 0, stream>>>(Xb, WqT, WkT, WvT, q_ws, k_ws, v_ws);
  }
  // 3) fused flash attention (WMMA + double-buffered async LDS staging)
  {
    dim3 grid(Tt / 64, Hh, Bb);
    flash_attn_kernel<<<grid, 128, 0, stream>>>(q_ws, k_ws, v_ws, o_ws);
  }
  // 4) output projection + bias (WMMA)
  {
    dim3 grid(BT / 16, Kk / 16);
    out_proj_kernel<<<grid, 32, 0, stream>>>(o_ws, WuT, bu, out);
  }
}
